// CenterAgent_69672959475832
// MI455X (gfx1250) — matrix operand order repaired and checked by reading, not compiled, not run on hardware
//
#include <hip/hip_runtime.h>
#include <hip/hip_bf16.h>
#include <math.h>

typedef __attribute__((ext_vector_type(16))) _Float16 v16h;
typedef __attribute__((ext_vector_type(8)))  _Float16 v8h;
typedef __attribute__((ext_vector_type(8)))  float    v8f;

#define HW   28
#define NPIX 784
#define BATCH 256

__device__ __forceinline__ v16h cat8(v8h lo, v8h hi) {
  return __builtin_shufflevector(lo, hi, 0,1,2,3,4,5,6,7,8,9,10,11,12,13,14,15);
}

// low 32 bits of a flat LDS pointer == LDS byte offset (aperture scheme)
__device__ __forceinline__ unsigned lds_u32(const void* p) {
  return (unsigned)(unsigned long long)p;
}

// async copy 16B global -> LDS, tracked by ASYNCcnt (CDNA5)
__device__ __forceinline__ void async_cp16(unsigned lds_addr, const void* gptr) {
  unsigned long long ga = (unsigned long long)gptr;
  asm volatile("global_load_async_to_lds_b128 %0, %1, off"
               :: "v"(lds_addr), "v"(ga) : "memory");
}

__device__ __forceinline__ void wait_async0() {
  asm volatile("s_wait_asynccnt 0x0" ::: "memory");
}

// ---------------- center map: one thread per batch, last-write-wins ----------------
__global__ void cmap_build(const float* __restrict__ centers, float* __restrict__ cmap) {
  int b = threadIdx.x;                       // 256 threads, 1 block
  float* row = cmap + (size_t)b * NPIX;
  for (int i = 0; i < NPIX; ++i) row[i] = 0.f;
  const float* cb = centers + (size_t)b * 100 * 4;
  for (int n = 0; n < 100; ++n) {
    int xp = (int)(cb[n*4+0] * 28.f);
    int yp = (int)(cb[n*4+1] * 28.f);
    float conf = cb[n*4+3];
    if (xp >= 0 && xp < HW && yp >= 0 && yp < HW) row[yp*HW + xp] = conf;
  }
}

// ---- fused concat input build: NHWC f16 [B][28][28][544], ch 0-2 image,
// ---- 3-514 bilinear-resized features (7->28, half-pixel, edge clamp), 515 cmap, 516-543 zero
__global__ void build_input(const float* __restrict__ image,      // [B][3][28][28]
                            const float* __restrict__ features,   // [B][512][7][7]
                            const float* __restrict__ cmap,       // [B][784]
                            _Float16* __restrict__ buf0)          // [B][784][544]
{
  const int pix = blockIdx.x, b = blockIdx.y, c = threadIdx.x;    // block=544 (17 waves)
  float v = 0.f;
  if (c < 3) {
    v = image[((size_t)b*3 + c)*NPIX + pix];
  } else if (c < 515) {
    const int fc = c - 3;
    const int y = pix / HW, x = pix % HW;
    float sy = y*0.25f - 0.375f;
    float sx = x*0.25f - 0.375f;
    int iy0 = (int)floorf(sy), ix0 = (int)floorf(sx);
    float wy = sy - (float)iy0, wx = sx - (float)ix0;
    int y0 = iy0 < 0 ? 0 : iy0;           int y1 = (iy0+1) > 6 ? 6 : (iy0+1 < 0 ? 0 : iy0+1);
    int x0 = ix0 < 0 ? 0 : ix0;           int x1 = (ix0+1) > 6 ? 6 : (ix0+1 < 0 ? 0 : ix0+1);
    const float* f = features + ((size_t)b*512 + fc)*49;
    float f00 = f[y0*7+x0], f01 = f[y0*7+x1], f10 = f[y1*7+x0], f11 = f[y1*7+x1];
    v = (1.f-wy)*((1.f-wx)*f00 + wx*f01) + wy*((1.f-wx)*f10 + wx*f11);
  } else if (c == 515) {
    v = cmap[(size_t)b*NPIX + pix];
  }
  buf0[((size_t)b*NPIX + pix)*544 + c] = (_Float16)v;
}

// ---- conv weight repack: fp32 [Cout][Cin][3][3] -> f16 [Cout][9][CINP] (zero-padded Cin)
__global__ void wprep(const float* __restrict__ w, _Float16* __restrict__ w16,
                      int Cout, int Cin, int CINP) {
  int idx = blockIdx.x * blockDim.x + threadIdx.x;
  int N = Cout * 9 * CINP;
  if (idx >= N) return;
  int cp   = idx % CINP;
  int tap  = (idx / CINP) % 9;
  int cout = idx / (9 * CINP);
  _Float16 v = (_Float16)0.f;
  if (cp < Cin) v = (_Float16)w[((size_t)cout*Cin + cp)*9 + tap];
  w16[idx] = v;
}

// ---- fc1 weight swizzle: fw1 fp32 [256][12544] (k = c*784 + p, NCHW flatten) ->
// ---- f16 [K/32][256][32] with NHWC k' = p*16 + c, B-fragment friendly
__global__ void fc1_swizzle(const float* __restrict__ fw1, _Float16* __restrict__ bsw) {
  int idx = blockIdx.x * blockDim.x + threadIdx.x;   // grid covers 392*256*32 exactly
  int kk  = idx & 31;
  int n   = (idx >> 5) & 255;
  int k32 = idx >> 13;
  int kprime = k32*32 + kk;
  int p = kprime >> 4, c = kprime & 15;
  bsw[idx] = (_Float16)fw1[(size_t)n*12544 + c*NPIX + p];
}

// ---------------- 3x3 SAME conv via implicit-GEMM WMMA ----------------
// in  : [B][28][28][KCH*32] f16   (NHWC, padded Cin)
// w16 : [COUT][9][KCH*32]   f16
// out : [B][28][28][COUT]   f16   (ReLU applied)
// grid (7, B), block 256.  WG covers 4 output rows (112 px = 7 N-tiles of 16).
// Double-buffered LDS staging via CDNA5 async global->LDS copies (ASYNCcnt).
template<int COUT, int KCH>
__launch_bounds__(256)
__global__ void conv3x3_wmma(const _Float16* __restrict__ in,
                             const _Float16* __restrict__ w16,
                             const float* __restrict__ bias,
                             _Float16* __restrict__ out)
{
  constexpr int CINP = KCH * 32;
  constexpr int MT   = COUT / 16;       // Cout tiles
  constexpr int NG   = 8 / MT;          // n-groups among 8 waves
  constexpr int MAXJ = (7 + NG - 1) / NG;
  __shared__ _Float16 smem[2][6 * 28 * 32];           // 2 x 10.5 KB, double buffered

  const int b = blockIdx.y;
  const int rowBase = blockIdx.x * 4;
  const int wave  = threadIdx.x >> 5;
  const int lane  = threadIdx.x & 31;
  const int mtile = wave % MT;
  const int ngrp  = wave / MT;
  const int lhalf = lane >> 4;          // 0/1
  const int l15   = lane & 15;

  // stage one 32-channel K-chunk (6 halo rows x 28 cols x 32 halves) into smem[sel]
  auto stage = [&](int kc, int sel) {
    for (int s = threadIdx.x; s < 168 * 4; s += 256) {
      int seg = s >> 2, part = s & 3;
      int lrow = seg / 28, col = seg - lrow * 28;
      int gy = rowBase - 1 + lrow;
      _Float16* dst = &smem[sel][seg*32 + part*8];
      if (gy >= 0 && gy < HW) {
        async_cp16(lds_u32(dst),
                   in + (((size_t)b*HW + gy)*HW + col)*CINP + kc*32 + part*8);
      } else {
        *(uint4*)dst = make_uint4(0u, 0u, 0u, 0u);    // halo rows -> zeros
      }
    }
  };

  v8f zacc = {};
  v8f acc[MAXJ];
#pragma unroll
  for (int jj = 0; jj < MAXJ; ++jj) acc[jj] = zacc;

  stage(0, 0);
  wait_async0();
  __syncthreads();

  for (int kc = 0; kc < KCH; ++kc) {
    const int cur = kc & 1;
    if (kc + 1 < KCH) stage(kc + 1, cur ^ 1);         // overlap next-chunk DMA with WMMA

#pragma unroll
    for (int tap = 0; tap < 9; ++tap) {
      const int dy = tap / 3, dx = tap % 3;
      // A fragment (weights): M = l15, K halves at {kb..kb+7} and {kb+16..kb+23}
      const _Float16* wr = w16 + ((size_t)(mtile*16 + l15)*9 + tap)*CINP + kc*32 + lhalf*8;
      v16h afrag = cat8(*(const v8h*)wr, *(const v8h*)(wr + 16));
#pragma unroll
      for (int jj = 0; jj < MAXJ; ++jj) {
        int j = ngrp + jj * NG;                 // wave-uniform
        if (j < 7) {
          int p  = j*16 + l15;                  // output pixel within 112-px strip
          int lr = p / 28 + dy;                 // staged row 0..5
          int ix = (p % 28) + dx - 1;
          v16h bfrag = {};
          if (ix >= 0 && ix < HW) {             // halo columns -> zeros
            const _Float16* lp = &smem[cur][(lr*28 + ix)*32 + lhalf*8];
            bfrag = cat8(*(const v8h*)lp, *(const v8h*)(lp + 16));
          }
          acc[jj] = __builtin_amdgcn_wmma_f32_16x16x32_f16(
              false, afrag, false, bfrag, (short)0, acc[jj], false, false);
        }
      }
    }

    wait_async0();          // this wave's async stage of kc+1 landed
    __syncthreads();        // everyone done reading smem[cur] + staging done
  }

  // epilogue: bias + ReLU, store 8 consecutive couts per lane (16B NHWC store)
#pragma unroll
  for (int jj = 0; jj < MAXJ; ++jj) {
    int j = ngrp + jj * NG;
    if (j < 7) {
      int p  = j*16 + l15;
      int oy = rowBase + p / 28, ox = p % 28;
      int cbase = mtile*16 + lhalf*8;
      v8h ov;
#pragma unroll
      for (int r = 0; r < 8; ++r) {
        float v = acc[jj][r] + bias[cbase + r];
        v = v > 0.f ? v : 0.f;
        ov[r] = (_Float16)v;
      }
      *(v8h*)(out + (((size_t)b*HW + oy)*HW + ox)*COUT + cbase) = ov;
    }
  }
}

// ---------------- FC1: [256,256] = relu(X[256,12544] @ W + b) via WMMA ----------------
__launch_bounds__(256)
__global__ void fc1_wmma(const _Float16* __restrict__ X,    // [256][12544] (NHWC-flat conv4 out)
                         const _Float16* __restrict__ bsw,  // [392][256][32]
                         const float* __restrict__ fb1,
                         _Float16* __restrict__ x1)         // [256][256] f16, ReLU'd
{
  const int mt = blockIdx.x;                 // 16 blocks: batch tiles
  const int wave = threadIdx.x >> 5, lane = threadIdx.x & 31;
  const int lhalf = lane >> 4, l15 = lane & 15;
  v8f zacc = {};
  v8f acc[2]; acc[0] = zacc; acc[1] = zacc;
  const _Float16* xrow = X + (size_t)(mt*16 + l15)*12544 + lhalf*8;
  for (int kc = 0; kc < 392; ++kc) {
    v16h a = cat8(*(const v8h*)(xrow + kc*32), *(const v8h*)(xrow + kc*32 + 16));
#pragma unroll
    for (int t = 0; t < 2; ++t) {
      int ncol = (wave*2 + t)*16 + l15;
      const _Float16* bp = bsw + ((size_t)kc*256 + ncol)*32 + lhalf*8;
      v16h bf = cat8(*(const v8h*)bp, *(const v8h*)(bp + 16));
      acc[t] = __builtin_amdgcn_wmma_f32_16x16x32_f16(
          false, a, false, bf, (short)0, acc[t], false, false);
    }
  }
#pragma unroll
  for (int t = 0; t < 2; ++t) {
    int col = (wave*2 + t)*16 + l15;
    float bv = fb1[col];
#pragma unroll
    for (int r = 0; r < 8; ++r) {
      int row = mt*16 + lhalf*8 + r;
      float v = acc[t][r] + bv;
      v = v > 0.f ? v : 0.f;
      x1[(size_t)row*256 + col] = (_Float16)v;
    }
  }
}

// ---------------- FC2 + sigmoid(cols 2,3): tiny, scalar ----------------
__global__ void fc2_final(const _Float16* __restrict__ x1, const float* __restrict__ fw2,
                          const float* __restrict__ fb2, float* __restrict__ out) {
  int b = blockIdx.x, o = threadIdx.x;       // 256 blocks x 32 threads
  if (o >= 24) return;
  float s = fb2[o];
  const _Float16* xr = x1 + (size_t)b*256;
  const float* wr = fw2 + (size_t)o*256;
  for (int k = 0; k < 256; ++k) s += (float)xr[k] * wr[k];
  if (o == 2 || o == 3) s = 1.f / (1.f + expf(-s));
  out[(size_t)b*24 + o] = s;
}

extern "C" void kernel_launch(void* const* d_in, const int* in_sizes, int n_in,
                              void* d_out, int out_size, void* d_ws, size_t ws_size,
                              hipStream_t stream) {
  const float* image    = (const float*)d_in[0];
  const float* features = (const float*)d_in[1];
  const float* centers  = (const float*)d_in[2];
  const float* w1 = (const float*)d_in[3];   const float* b1 = (const float*)d_in[4];
  const float* w2 = (const float*)d_in[5];   const float* b2 = (const float*)d_in[6];
  const float* w3 = (const float*)d_in[7];   const float* b3 = (const float*)d_in[8];
  const float* w4 = (const float*)d_in[9];   const float* b4 = (const float*)d_in[10];
  const float* fw1 = (const float*)d_in[11]; const float* fb1 = (const float*)d_in[12];
  const float* fw2 = (const float*)d_in[13]; const float* fb2 = (const float*)d_in[14];

  char* ws = (char*)d_ws;
  size_t off = 0;
  auto alloc = [&](size_t bytes) -> void* {
    void* p = ws + off;
    off += (bytes + 255) & ~(size_t)255;
    return p;
  };
  _Float16* buf0 = (_Float16*)alloc((size_t)BATCH*NPIX*544*2);  // concat input, NHWC padded
  _Float16* buf1 = (_Float16*)alloc((size_t)BATCH*NPIX*128*2);
  _Float16* buf2 = (_Float16*)alloc((size_t)BATCH*NPIX*64*2);
  _Float16* buf3 = (_Float16*)alloc((size_t)BATCH*NPIX*32*2);
  _Float16* buf4 = (_Float16*)alloc((size_t)BATCH*NPIX*16*2);   // == X [256][12544]
  float*    cmap = (float*)   alloc((size_t)BATCH*NPIX*4);
  _Float16* w16a = (_Float16*)alloc((size_t)128*9*544*2);
  _Float16* w16b = (_Float16*)alloc((size_t)64*9*128*2);
  _Float16* w16c = (_Float16*)alloc((size_t)32*9*64*2);
  _Float16* w16d = (_Float16*)alloc((size_t)16*9*32*2);
  _Float16* bsw1 = (_Float16*)alloc((size_t)392*256*32*2);
  _Float16* x1   = (_Float16*)alloc((size_t)256*256*2);
  (void)ws_size; (void)in_sizes; (void)n_in; (void)out_size;

  // prep
  cmap_build<<<1, 256, 0, stream>>>(centers, cmap);
  build_input<<<dim3(NPIX, BATCH), 544, 0, stream>>>(image, features, cmap, buf0);
  wprep<<<(128*9*544 + 255)/256, 256, 0, stream>>>(w1, w16a, 128, 516, 544);
  wprep<<<( 64*9*128 + 255)/256, 256, 0, stream>>>(w2, w16b,  64, 128, 128);
  wprep<<<( 32*9* 64 + 255)/256, 256, 0, stream>>>(w3, w16c,  32,  64,  64);
  wprep<<<( 16*9* 32 + 255)/256, 256, 0, stream>>>(w4, w16d,  16,  32,  32);
  fc1_swizzle<<<(392*256*32)/256, 256, 0, stream>>>(fw1, bsw1);

  // conv stack (WMMA, async double-buffered LDS staging)
  conv3x3_wmma<128,17><<<dim3(7, BATCH), 256, 0, stream>>>(buf0, w16a, b1, buf1);
  conv3x3_wmma< 64, 4><<<dim3(7, BATCH), 256, 0, stream>>>(buf1, w16b, b2, buf2);
  conv3x3_wmma< 32, 2><<<dim3(7, BATCH), 256, 0, stream>>>(buf2, w16c, b3, buf3);
  conv3x3_wmma< 16, 1><<<dim3(7, BATCH), 256, 0, stream>>>(buf3, w16d, b4, buf4);

  // FCs
  fc1_wmma<<<16, 256, 0, stream>>>(buf4, bsw1, fb1, x1);
  fc2_final<<<BATCH, 32, 0, stream>>>(x1, fw2, fb2, (float*)d_out);
}
